// Attention_38276748542802
// MI455X (gfx1250) — matrix operand order repaired and checked by reading, not compiled, not run on hardware
//
#include <hip/hip_runtime.h>
#include <math.h>

typedef __attribute__((ext_vector_type(16))) _Float16 v16h;
typedef __attribute__((ext_vector_type(8)))  float    v8f;

#define B_   128
#define S_   1024
#define EH_  512
#define DH_  256

// ---------------------------------------------------------------------------
// Kernel 1: pack W_e = attn_w[256:768, :] (f32, [512,256] row-major) into f16,
// pre-swizzled into a contiguous-k WMMA 16x16x32 B-fragment per-lane layout:
//   fragment (d_tile j, k_step t), lane L:  column N = L&15  (d = j*16+N)
//   16 halfs = K values (t*32 + (L>=16 ? 16 : 0)) + 0..15, consecutive.
// Stored per-lane contiguous (32B) so the hot loop does one v16h load/lane.
// ---------------------------------------------------------------------------
__global__ void pack_we(const float* __restrict__ attn_w,
                        _Float16* __restrict__ wpack) {
    int id   = blockIdx.x * blockDim.x + threadIdx.x;   // 0..8191
    int lane = id & 31;
    int t    = (id >> 5) & 15;          // k-step (K=512 / 32)
    int j    = id >> 9;                 // d-tile (256 / 16)
    int d    = j * 16 + (lane & 15);
    int kb   = t * 32 + ((lane >> 4) << 4);
    v16h frag;
#pragma unroll
    for (int i = 0; i < 16; ++i) {
        int k = kb + i;
        frag[i] = (_Float16)attn_w[(size_t)(DH_ + k) * DH_ + d];
    }
    *reinterpret_cast<v16h*>(wpack + (size_t)id * 16) = frag;
}

// ---------------------------------------------------------------------------
// Kernel 2: hp[b,d] = attn_b[d] + hidden[b,:] @ attn_w[:256, d]   (tiny)
// ---------------------------------------------------------------------------
__global__ void h_part(const float* __restrict__ hidden,
                       const float* __restrict__ attn_w,
                       const float* __restrict__ attn_b,
                       float* __restrict__ hp) {
    __shared__ float hrow[DH_];
    int b = blockIdx.x, d = threadIdx.x;
    hrow[d] = hidden[b * DH_ + d];
    __syncthreads();
    float acc = attn_b[d];
#pragma unroll 8
    for (int k = 0; k < DH_; ++k)
        acc = fmaf(hrow[k], attn_w[(size_t)k * DH_ + d], acc);
    hp[b * DH_ + d] = acc;
}

// ---------------------------------------------------------------------------
// Kernel 3: fused  e_part GEMM -> tanh -> dot(v) -> mask  for one (s, b-tile).
// Grid (8, 1024); 256 threads = 8 wave32; wave w owns d-tiles {2w, 2w+1}.
// A fragment (16x32 f16) built on the fly from f32 encoder rows:
//   lane L: row M = L&15; K offsets = t*32 + (L<16 ? {0..15}
//                                                  : {16..31}).
// C layout: VGPR i -> row i (lanes 0-15, N=lane) / row i+8 (lanes 16-31).
// ---------------------------------------------------------------------------
__global__ void __launch_bounds__(256)
attn_scores(const float* __restrict__ enc,
            const _Float16* __restrict__ wpack,
            const float* __restrict__ hp,
            const float* __restrict__ v,
            const unsigned char* __restrict__ mask,
            float* __restrict__ scores) {
    __shared__ float hp_t[16 * DH_];   // 16 KB: hp tile for this b-tile
    __shared__ float vbuf[DH_];        //  1 KB
    __shared__ float red[16];          // per-row score accumulators

    const int tid = threadIdx.x;
    const int b0  = blockIdx.x * 16;
    const int s   = blockIdx.y;

    // Stage hp tile, v, zero reduction buffer.
#pragma unroll
    for (int idx = tid; idx < 16 * DH_; idx += 256)
        hp_t[idx] = hp[(b0 + (idx >> 8)) * DH_ + (idx & (DH_ - 1))];
    vbuf[tid] = v[tid];
    if (tid < 16) red[tid] = 0.0f;
    __syncthreads();

    const int lane = tid & 31;
    const int wave = tid >> 5;
    const int j0 = wave * 2, j1 = j0 + 1;
    const int r    = lane & 15;
    const int hi   = lane >> 4;            // 0 = low half-wave, 1 = high
    const float* arow = enc + ((size_t)s * B_ + b0 + r) * EH_;
    const v16h*  wp   = reinterpret_cast<const v16h*>(wpack);

    v8f acc0 = {};
    v8f acc1 = {};

#pragma unroll 4
    for (int t = 0; t < 16; ++t) {         // K = 512, 32 per step
        const int kb = t * 32 + hi * 16;
        float4 x0 = *reinterpret_cast<const float4*>(arow + kb);
        float4 x1 = *reinterpret_cast<const float4*>(arow + kb + 4);
        float4 x2 = *reinterpret_cast<const float4*>(arow + kb + 8);
        float4 x3 = *reinterpret_cast<const float4*>(arow + kb + 12);
        if (t < 15) __builtin_prefetch(arow + kb + 32, 0, 0); // next k-step

        v16h a;
        a[0]  = (_Float16)x0.x; a[1]  = (_Float16)x0.y;
        a[2]  = (_Float16)x0.z; a[3]  = (_Float16)x0.w;
        a[4]  = (_Float16)x1.x; a[5]  = (_Float16)x1.y;
        a[6]  = (_Float16)x1.z; a[7]  = (_Float16)x1.w;
        a[8]  = (_Float16)x2.x; a[9]  = (_Float16)x2.y;
        a[10] = (_Float16)x2.z; a[11] = (_Float16)x2.w;
        a[12] = (_Float16)x3.x; a[13] = (_Float16)x3.y;
        a[14] = (_Float16)x3.z; a[15] = (_Float16)x3.w;

        v16h bf0 = wp[(size_t)(j0 * 16 + t) * 32 + lane];
        v16h bf1 = wp[(size_t)(j1 * 16 + t) * 32 + lane];

        acc0 = __builtin_amdgcn_wmma_f32_16x16x32_f16(
                   false, a, false, bf0, (short)0, acc0, false, false);
        acc1 = __builtin_amdgcn_wmma_f32_16x16x32_f16(
                   false, a, false, bf1, (short)0, acc1, false, false);
    }

    // Epilogue: tanh(hp + e) * v, reduce over d, accumulate per batch row.
    const int n      = lane & 15;
    const int rowoff = hi * 8;
#pragma unroll
    for (int i = 0; i < 8; ++i) {
        const int row = i + rowoff;
        const int d0 = j0 * 16 + n, d1 = j1 * 16 + n;
        float ts = tanhf(hp_t[row * DH_ + d0] + acc0[i]) * vbuf[d0]
                 + tanhf(hp_t[row * DH_ + d1] + acc1[i]) * vbuf[d1];
        // reduce over the 16 N-lanes of each half-wave
        ts += __shfl_xor(ts, 1, 32);
        ts += __shfl_xor(ts, 2, 32);
        ts += __shfl_xor(ts, 4, 32);
        ts += __shfl_xor(ts, 8, 32);
        if ((lane & 15) == 0) atomicAdd(&red[row], ts);   // ds_add_f32
    }
    __syncthreads();

    if (tid < 16) {
        const int b = b0 + tid;
        float sc = red[tid];
        if (mask[(size_t)b * S_ + s]) sc = -1000000.0f;
        scores[(size_t)b * S_ + s] = sc;
    }
}

// ---------------------------------------------------------------------------
// Kernel 4: softmax over S per batch row, in place. One wave32 per row.
// ---------------------------------------------------------------------------
__global__ void softmax_rows(float* __restrict__ out) {
    const int b = blockIdx.x;
    const int lane = threadIdx.x;            // 32 threads
    float* row = out + (size_t)b * S_;
    float vals[S_ / 32];
    float m = -3.4e38f;
#pragma unroll
    for (int i = 0; i < S_ / 32; ++i) {
        vals[i] = row[lane + i * 32];
        m = fmaxf(m, vals[i]);
    }
    for (int off = 16; off; off >>= 1) m = fmaxf(m, __shfl_xor(m, off, 32));
    float sum = 0.0f;
#pragma unroll
    for (int i = 0; i < S_ / 32; ++i) {
        vals[i] = __expf(vals[i] - m);
        sum += vals[i];
    }
    for (int off = 16; off; off >>= 1) sum += __shfl_xor(sum, off, 32);
    const float inv = 1.0f / sum;
#pragma unroll
    for (int i = 0; i < S_ / 32; ++i) row[lane + i * 32] = vals[i] * inv;
}

// ---------------------------------------------------------------------------
extern "C" void kernel_launch(void* const* d_in, const int* in_sizes, int n_in,
                              void* d_out, int out_size, void* d_ws, size_t ws_size,
                              hipStream_t stream) {
    (void)in_sizes; (void)n_in; (void)out_size; (void)ws_size;
    const float*         hidden = (const float*)d_in[0];         // [128,256]
    const float*         enc    = (const float*)d_in[1];         // [1024,128,512]
    const unsigned char* mask   = (const unsigned char*)d_in[2]; // [128,1024] bool (1B)
    const float*         attn_w = (const float*)d_in[3];         // [768,256]
    const float*         attn_b = (const float*)d_in[4];         // [256]
    const float*         v      = (const float*)d_in[5];         // [256]
    float*               out    = (float*)d_out;                 // [128,1024]

    // Workspace: 256 KB f16 W-pack + 128 KB hp at +512 KB (both rebuilt every call).
    _Float16* wpack = (_Float16*)d_ws;
    float*    hp    = (float*)((char*)d_ws + 512 * 1024);

    pack_we<<<32, 256, 0, stream>>>(attn_w, wpack);
    h_part<<<128, 256, 0, stream>>>(hidden, attn_w, attn_b, hp);
    dim3 grid(B_ / 16, S_);
    attn_scores<<<grid, 256, 0, stream>>>(enc, wpack, hp, v, mask, out);
    softmax_rows<<<B_, 32, 0, stream>>>(out);
}
